// NeuronCircuit_15229954031678
// MI455X (gfx1250) — matrix operand order repaired and checked by reading, not compiled, not run on hardware
//
#include <hip/hip_runtime.h>
#include <hip/hip_bf16.h>

// ---------------------------------------------------------------------------
// Types
// ---------------------------------------------------------------------------
typedef __bf16 bf16_t;
typedef __attribute__((ext_vector_type(16))) __bf16 v16bf;
typedef __attribute__((ext_vector_type(8)))  __bf16 v8bf;
typedef __attribute__((ext_vector_type(8)))  float   v8f;

#define BB   4
#define SS   2048
#define DDIM 1024
#define HH   16
#define RR   64
#define NNEU 16
#define DH   64
#define TOK  (BB * SS)   // 8192 tokens

// ---------------------------------------------------------------------------
// CDNA5 async memory->LDS copy (GLOBAL_LOAD_ASYNC_TO_LDS_B128, ASYNCcnt).
// GVS addressing: mem = SGPR64 base + VGPR32 byte offset. The LDS destination
// operand is a per-lane LDS byte address: low 32 bits of the generic pointer
// to __shared__ (LDS aperture truncation, ISA 10.2).
// ---------------------------------------------------------------------------
static __device__ __forceinline__ void async_b128(void* lds_dst,
                                                  const void* gbase,
                                                  unsigned byte_off) {
  unsigned l = (unsigned)(unsigned long long)lds_dst;
  unsigned long long s = (unsigned long long)gbase;
  asm volatile("global_load_async_to_lds_b128 %0, %1, %2"
               :: "v"(l), "v"(byte_off), "s"(s)
               : "memory");
}

static __device__ __forceinline__ void wait_async0() {
  asm volatile("s_wait_asynccnt 0" ::: "memory");
}

// ---------------------------------------------------------------------------
// VALU 16-lane XOR butterfly (v_permlane16_b32): lane j reads lane j^m within
// each 16-lane group. Replaces ds_bpermute + s_wait_dscnt round trips.
// ---------------------------------------------------------------------------
static __device__ __forceinline__ float permlane16_xor(float v, int lo, int hi) {
  int iv = __float_as_int(v);
  int r  = __builtin_amdgcn_permlane16(iv, iv, lo, hi, false, false);
  return __int_as_float(r);
}

static __device__ __forceinline__ float half_reduce_max(float v) {
  v = fmaxf(v, permlane16_xor(v, 0x67452301, 0xEFCDAB89));   // m=1
  v = fmaxf(v, permlane16_xor(v, 0x54761032, 0xDCFE98BA));   // m=2
  v = fmaxf(v, permlane16_xor(v, 0x32107654, 0xBA98FEDC));   // m=4
  v = fmaxf(v, permlane16_xor(v, 0xFEDCBA98, 0x76543210));   // m=8
  return v;
}

static __device__ __forceinline__ float half_reduce_sum(float v) {
  v += permlane16_xor(v, 0x67452301, 0xEFCDAB89);            // m=1
  v += permlane16_xor(v, 0x54761032, 0xDCFE98BA);            // m=2
  v += permlane16_xor(v, 0x32107654, 0xBA98FEDC);            // m=4
  v += permlane16_xor(v, 0xFEDCBA98, 0x76543210);            // m=8
  return v;
}

// ---------------------------------------------------------------------------
// WMMA fragment helpers (wave32, v_wmma_f32_16x16x32_bf16)
//
// A fragment (16x32, row-major source, ld = row stride in elements):
//   lane 0..15  -> M = lane,     K = {0..7, 16..23}
//   lane 16..31 -> M = lane-16,  K = {8..15, 24..31}
// B fragment (32x16, row-major source B[k][n]):
//   lane holds column n = lane&15; K = {8*half+0..7, 16+8*half+0..7} (same as A)
// C/D (16x16 f32): VGPR i -> row (i + 8*(lane>>4)), col = lane&15
// ---------------------------------------------------------------------------
static __device__ __forceinline__ v8f v8f_zero() {
  v8f z = {0.f, 0.f, 0.f, 0.f, 0.f, 0.f, 0.f, 0.f};
  return z;
}

static __device__ __forceinline__ v16bf frag_a_load(const bf16_t* base, int ld) {
  const int lane = threadIdx.x & 31;
  const int half = lane >> 4;
  const int m    = lane & 15;
  const bf16_t* p = base + (size_t)m * ld + 8 * half;
  v16bf f;
  *(v8bf*)&f         = *(const v8bf*)p;         // K = 8*half + 0..7
  *((v8bf*)&f + 1)   = *(const v8bf*)(p + 16);  // K = 16 + 8*half + 0..7
  return f;
}

static __device__ __forceinline__ v16bf frag_b_load(const bf16_t* base, int ld) {
  const int lane = threadIdx.x & 31;
  const int half = lane >> 4;
  const int n    = lane & 15;
  const bf16_t* p = base + (size_t)(8 * half) * ld + n;
  v16bf f;
#pragma unroll
  for (int e = 0; e < 8; ++e) {
    f[e]     = p[(size_t)e * ld];         // K = 8*half + e
    f[8 + e] = p[(size_t)(16 + e) * ld];  // K = 16 + 8*half + e
  }
  return f;
}

template <typename T>
static __device__ __forceinline__ void store_tile(T* out, int ld, const v8f& c) {
  const int lane = threadIdx.x & 31;
  const int half = lane >> 4;
  const int n    = lane & 15;
#pragma unroll
  for (int i = 0; i < 8; ++i)
    out[(size_t)(i + 8 * half) * ld + n] = (T)c[i];
}

static __device__ __forceinline__ v8f wmma_bf16(const v16bf& a, const v16bf& b, const v8f& c) {
  return __builtin_amdgcn_wmma_f32_16x16x32_bf16(false, a, false, b, (short)0, c,
                                                 false, false);
}

// ---------------------------------------------------------------------------
// Prep kernels: fp32 -> bf16 conversions / repacks
// ---------------------------------------------------------------------------
__global__ void k_cvt_bf16(const float* __restrict__ in, bf16_t* __restrict__ out, int n) {
  int i = blockIdx.x * blockDim.x + threadIdx.x;
  if (i < n) out[i] = (bf16_t)in[i];
}

// compress_neurons [N][D][R] -> Cp[d][n*R + r]  (so compress is one GEMM)
__global__ void k_pack_compress(const float* __restrict__ C, bf16_t* __restrict__ Cp) {
  int i = blockIdx.x * blockDim.x + threadIdx.x;  // over N*D*R = 1M
  if (i >= NNEU * DDIM * RR) return;
  int r = i & (RR - 1);
  int d = (i >> 6) & (DDIM - 1);
  int n = i >> 16;
  Cp[(size_t)d * (NNEU * RR) + n * RR + r] = (bf16_t)C[i];
}

// W_O [dout][din] -> WOt[din][dout]  (y = attn @ W_O^T)
__global__ void k_pack_wo(const float* __restrict__ W, bf16_t* __restrict__ Wt) {
  int i = blockIdx.x * blockDim.x + threadIdx.x;  // over D*D
  if (i >= DDIM * DDIM) return;
  int din  = i & (DDIM - 1);
  int dout = i >> 10;
  Wt[(size_t)din * DDIM + dout] = (bf16_t)W[i];
}

// ---------------------------------------------------------------------------
// Tiled bf16 GEMM: C[M,Nc] = A[M,K] @ B[K,Nc], fp32 accumulate.
// Block tile 64x128, K-step 32. 256 threads = 8 waves; wave (wm,wn) owns a
// 16x64 strip -> 4 v8f accumulators. Tiles staged with async->LDS DMA.
// M%64==0, Nc%128==0, K%32==0 assumed.
// ---------------------------------------------------------------------------
template <typename OutT>
__global__ __launch_bounds__(256) void k_gemm_bf16(const bf16_t* __restrict__ A,
                                                   const bf16_t* __restrict__ Bm,
                                                   OutT* __restrict__ Cmat,
                                                   int M, int Nc, int K) {
  __shared__ __align__(16) bf16_t As[64 * 32];
  __shared__ __align__(16) bf16_t Bs[32 * 128];

  const int bm = blockIdx.x;        // M/64 tiles
  const int bn = blockIdx.y;        // Nc/128 tiles
  const int wave = threadIdx.x >> 5;
  const int wm = wave & 3;          // 0..3 : 16-row strip
  const int wn = wave >> 2;         // 0..1 : 64-col strip

  v8f acc[4];
#pragma unroll
  for (int j = 0; j < 4; ++j) acc[j] = v8f_zero();

  const int t = threadIdx.x;
  const int arow = t >> 2, acol = (t & 3) * 8;     // A tile:  64x32, 8 elems/thread
  const int brow = t >> 3, bcol = (t & 7) * 16;    // B tile:  32x128, 16 elems/thread

  // per-thread global byte offsets (advance by K-step each iteration)
  unsigned aoff = (unsigned)(((size_t)(bm * 64 + arow) * K + acol) * sizeof(bf16_t));
  unsigned boff = (unsigned)(((size_t)brow * Nc + bn * 128 + bcol) * sizeof(bf16_t));

  for (int k0 = 0; k0 < K; k0 += 32) {
    // async DMA the A/B slabs into LDS (no VGPR round trip, ASYNCcnt tracked)
    async_b128(&As[arow * 32 + acol],      A,  aoff);
    async_b128(&Bs[brow * 128 + bcol],     Bm, boff);
    async_b128(&Bs[brow * 128 + bcol + 8], Bm, boff + 16u);
    aoff += 32u * sizeof(bf16_t);
    boff += (unsigned)(32u * Nc * sizeof(bf16_t));

    wait_async0();
    __syncthreads();

    v16bf a = frag_a_load(&As[wm * 16 * 32], 32);
#pragma unroll
    for (int j = 0; j < 4; ++j) {
      v16bf b = frag_b_load(&Bs[wn * 64 + j * 16], 128);
      acc[j] = wmma_bf16(a, b, acc[j]);
    }
    __syncthreads();
  }

#pragma unroll
  for (int j = 0; j < 4; ++j) {
    OutT* out = &Cmat[(size_t)(bm * 64 + wm * 16) * Nc + bn * 128 + wn * 64 + j * 16];
    store_tile(out, Nc, acc[j]);
  }
}

// ---------------------------------------------------------------------------
// Mix: h[t,r] = sum_n cw[t,n] * Xc[t, n*R+r];
//      HwQ[t, n*R+r] = bf16(ewq[t,n] * h[t,r]),  HwV likewise.
// One 64-thread block per token (thread = r).
// ---------------------------------------------------------------------------
__global__ __launch_bounds__(64) void k_mix(const float* __restrict__ Xc,
                                            const float* __restrict__ cw,
                                            const float* __restrict__ ewq,
                                            const float* __restrict__ ewv,
                                            bf16_t* __restrict__ HwQ,
                                            bf16_t* __restrict__ HwV) {
  const int tk = blockIdx.x;
  const int r  = threadIdx.x;
  __shared__ float s_c[NNEU], s_q[NNEU], s_v[NNEU];
  if (threadIdx.x < NNEU) {
    s_c[threadIdx.x] = cw [tk * NNEU + threadIdx.x];
    s_q[threadIdx.x] = ewq[tk * NNEU + threadIdx.x];
    s_v[threadIdx.x] = ewv[tk * NNEU + threadIdx.x];
  }
  __syncthreads();
  float h = 0.f;
#pragma unroll
  for (int n = 0; n < NNEU; ++n) h += s_c[n] * Xc[(size_t)tk * DDIM + n * RR + r];
#pragma unroll
  for (int n = 0; n < NNEU; ++n) {
    HwQ[(size_t)tk * DDIM + n * RR + r] = (bf16_t)(s_q[n] * h);
    HwV[(size_t)tk * DDIM + n * RR + r] = (bf16_t)(s_v[n] * h);
  }
}

// ---------------------------------------------------------------------------
// Causal flash attention. K == Q (reference computes identical tensors).
// Grid: (S/64, H, B); 128 threads = 4 waves, wave w owns query rows
// qbase + w*16 .. +15.  KV tiles of 32 keys.  fp32 softmax state per row.
// V tile staged with async->LDS DMA; K tile needs a transpose so it is
// staged with explicit scatter stores. Row reductions use v_permlane16.
// ---------------------------------------------------------------------------
__global__ __launch_bounds__(128) void k_attn(const bf16_t* __restrict__ Qb,
                                              const bf16_t* __restrict__ Vb,
                                              bf16_t* __restrict__ Ob) {
  const int qt = blockIdx.x, h = blockIdx.y, b = blockIdx.z;
  const int wave = threadIdx.x >> 5;
  const int lane = threadIdx.x & 31;
  const int half = lane >> 4;
  const int ln   = lane & 15;
  const int qbase = qt * 64;

  const bf16_t* Qh = Qb + (size_t)b * SS * DDIM + h * DH;  // row stride DDIM
  const bf16_t* Vh = Vb + (size_t)b * SS * DDIM + h * DH;

  __shared__ __align__(16) bf16_t KT[64 * 32];      // [d][key] (transposed)
  __shared__ __align__(16) bf16_t Vs[32 * 64];      // [key][d]
  __shared__ __align__(16) bf16_t Ps[4][16 * 32];   // per-wave P staging

  // Q fragments (2 K-chunks of 32 over DH=64), loaded straight from global
  const int qrow = qbase + wave * 16 + ln;          // A-frag row for this lane
  v16bf qf[2];
#pragma unroll
  for (int c = 0; c < 2; ++c) {
    const bf16_t* p = Qh + (size_t)qrow * DDIM + c * 32 + 8 * half;
    *(v8bf*)&qf[c]       = *(const v8bf*)p;
    *((v8bf*)&qf[c] + 1) = *(const v8bf*)(p + 16);
  }

  float mrow[8], lrow[8];
  v8f oacc[4];
#pragma unroll
  for (int i = 0; i < 8; ++i) { mrow[i] = -1e30f; lrow[i] = 0.f; }
#pragma unroll
  for (int j = 0; j < 4; ++j) oacc[j] = v8f_zero();

  const int key_ld = threadIdx.x >> 2;        // 0..31 (loader key index)
  const int dc     = (threadIdx.x & 3) * 16;  // 0,16,32,48
  const float scale = 0.125f;                 // 1/sqrt(64)
  const int nkv = qt * 2 + 2;

  for (int kt = 0; kt < nkv; ++kt) {
    const int kbase = kt * 32;
    __syncthreads();  // previous tiles fully consumed
    {
      // V tile: straight copy -> async DMA (2 x 16B per thread)
      unsigned voff =
          (unsigned)(((size_t)(kbase + key_ld) * DDIM + dc) * sizeof(bf16_t));
      async_b128(&Vs[key_ld * 64 + dc],     Vh, voff);
      async_b128(&Vs[key_ld * 64 + dc + 8], Vh, voff + 16u);

      // K tile (= Q) needs d-major transpose: load + scatter into LDS
      const bf16_t* src = Qh + (size_t)(kbase + key_ld) * DDIM + dc;
      v8bf k0 = *(const v8bf*)src;
      v8bf k1 = *(const v8bf*)(src + 8);
#pragma unroll
      for (int e = 0; e < 8; ++e) {
        KT[(dc + e) * 32 + key_ld]     = k0[e];
        KT[(dc + 8 + e) * 32 + key_ld] = k1[e];
      }
    }
    wait_async0();
    __syncthreads();

    // S = Q Kt : two 16x16 score subtiles, each K-accumulated over DH
    v8f sacc[2];
#pragma unroll
    for (int nt = 0; nt < 2; ++nt) {
      sacc[nt] = v8f_zero();
#pragma unroll
      for (int c = 0; c < 2; ++c) {
        v16bf kf = frag_b_load(&KT[(c * 32) * 32 + nt * 16], 32);
        sacc[nt] = wmma_bf16(qf[c], kf, sacc[nt]);
      }
    }

    // scale + causal mask + online softmax (row-wise across 16 lanes per half)
#pragma unroll
    for (int i = 0; i < 8; ++i) {
      const int qr = qbase + wave * 16 + i + 8 * half;
      float mx = -1e30f;
#pragma unroll
      for (int nt = 0; nt < 2; ++nt) {
        const int kc = kbase + nt * 16 + ln;
        float s = sacc[nt][i] * scale;
        if (kc > qr) s = -1e30f;
        sacc[nt][i] = s;
        mx = fmaxf(mx, s);
      }
      mx = half_reduce_max(mx);
      const float mnew  = fmaxf(mrow[i], mx);
      const float alpha = __expf(mrow[i] - mnew);
      float rs = 0.f;
#pragma unroll
      for (int nt = 0; nt < 2; ++nt) {
        float p = __expf(sacc[nt][i] - mnew);
        sacc[nt][i] = p;
        rs += p;
      }
      rs = half_reduce_sum(rs);
      lrow[i] = lrow[i] * alpha + rs;
      mrow[i] = mnew;
#pragma unroll
      for (int j = 0; j < 4; ++j) oacc[j][i] *= alpha;
    }

    // C-layout P -> A-layout via per-wave LDS staging
#pragma unroll
    for (int i = 0; i < 8; ++i)
#pragma unroll
      for (int nt = 0; nt < 2; ++nt)
        Ps[wave][(i + 8 * half) * 32 + nt * 16 + ln] = (bf16_t)sacc[nt][i];

    v16bf pf = frag_a_load(&Ps[wave][0], 32);
#pragma unroll
    for (int j = 0; j < 4; ++j) {
      v16bf vf = frag_b_load(&Vs[j * 16], 64);
      oacc[j] = wmma_bf16(pf, vf, oacc[j]);
    }
  }

  // epilogue: normalize and store (output layout [token][h*64+dh])
#pragma unroll
  for (int j = 0; j < 4; ++j)
#pragma unroll
    for (int i = 0; i < 8; ++i) {
      const int row = qbase + wave * 16 + i + 8 * half;
      Ob[(size_t)b * SS * DDIM + (size_t)row * DDIM + h * DH + j * 16 + ln] =
          (bf16_t)(oacc[j][i] / lrow[i]);
    }
}

// ---------------------------------------------------------------------------
// Host launcher
// ---------------------------------------------------------------------------
extern "C" void kernel_launch(void* const* d_in, const int* in_sizes, int n_in,
                              void* d_out, int out_size, void* d_ws, size_t ws_size,
                              hipStream_t stream) {
  (void)in_sizes; (void)n_in; (void)out_size; (void)ws_size;

  const float* x   = (const float*)d_in[0];
  const float* cw  = (const float*)d_in[1];
  const float* ewq = (const float*)d_in[2];
  const float* ewv = (const float*)d_in[3];
  const float* Cn  = (const float*)d_in[4];
  const float* Eq  = (const float*)d_in[5];
  const float* Ev  = (const float*)d_in[6];
  const float* Wo  = (const float*)d_in[7];

  char* w = (char*)d_ws;
  const size_t MB = 1u << 20;
  bf16_t* Xh   = (bf16_t*)(w + 0 * MB);    // 16 MB  x in bf16
  bf16_t* Cp   = (bf16_t*)(w + 16 * MB);   //  2 MB  compress repacked [D, N*R]
  bf16_t* EQKh = (bf16_t*)(w + 18 * MB);   //  2 MB  expand QK  [N*R, D]
  bf16_t* EVh  = (bf16_t*)(w + 20 * MB);   //  2 MB  expand V   [N*R, D]
  bf16_t* WOt  = (bf16_t*)(w + 22 * MB);   //  2 MB  W_O^T
  bf16_t* HwQ  = (bf16_t*)(w + 24 * MB);   // 16 MB  (ew_QK * h)
  bf16_t* HwV  = (bf16_t*)(w + 40 * MB);   // 16 MB  (ew_V  * h)
  float*  Xc   = (float *)(w + 56 * MB);   // 32 MB  X @ Cp
  bf16_t* Qb   = (bf16_t*)(w + 56 * MB);   // 16 MB  (reuses Xc after mix)
  bf16_t* Vb   = (bf16_t*)(w + 72 * MB);   // 16 MB
  bf16_t* At   = (bf16_t*)(w + 24 * MB);   // 16 MB  attn out (reuses HwQ)

  const int THR = 256;
  // prep / repack
  k_cvt_bf16<<<(TOK * DDIM + THR - 1) / THR, THR, 0, stream>>>(x, Xh, TOK * DDIM);
  k_pack_compress<<<(NNEU * DDIM * RR + THR - 1) / THR, THR, 0, stream>>>(Cn, Cp);
  k_cvt_bf16<<<(NNEU * RR * DDIM + THR - 1) / THR, THR, 0, stream>>>(Eq, EQKh, NNEU * RR * DDIM);
  k_cvt_bf16<<<(NNEU * RR * DDIM + THR - 1) / THR, THR, 0, stream>>>(Ev, EVh, NNEU * RR * DDIM);
  k_pack_wo<<<(DDIM * DDIM + THR - 1) / THR, THR, 0, stream>>>(Wo, WOt);

  dim3 gg(TOK / 64, DDIM / 128);
  // compress GEMM: Xc = Xh @ Cp
  k_gemm_bf16<float><<<gg, 256, 0, stream>>>(Xh, Cp, Xc, TOK, DDIM, DDIM);
  // neuron mixing
  k_mix<<<TOK, 64, 0, stream>>>(Xc, cw, ewq, ewv, HwQ, HwV);
  // expand GEMMs: Q(=K) and V
  k_gemm_bf16<bf16_t><<<gg, 256, 0, stream>>>(HwQ, EQKh, Qb, TOK, DDIM, DDIM);
  k_gemm_bf16<bf16_t><<<gg, 256, 0, stream>>>(HwV, EVh, Vb, TOK, DDIM, DDIM);
  // causal flash attention
  k_attn<<<dim3(SS / 64, HH, BB), 128, 0, stream>>>(Qb, Vb, At);
  // output projection: out = At @ W_O^T
  k_gemm_bf16<float><<<gg, 256, 0, stream>>>(At, WOt, (float*)d_out, TOK, DDIM, DDIM);
}
